// FullGraphConvolution_72206990181161
// MI455X (gfx1250) — compile-verified, matrix operand order checked by reading, not run on hardware
//
#include <hip/hip_runtime.h>

#define N_NODES 1024
#define F_IN    128
#define KTOP    9
#define OUT_CH  256
#define KF      (KTOP * F_IN)   // 1152 contraction length

typedef __attribute__((ext_vector_type(16))) _Float16 v16h;
typedef __attribute__((ext_vector_type(8)))  _Float16 v8h;
typedef __attribute__((ext_vector_type(8)))  float    v8f;

// ---------------------------------------------------------------------------
// Kernel 0: adjT[j][i] = adj[i][j]   (LDS-tiled transpose, 32x32 tiles)
// ---------------------------------------------------------------------------
__global__ __launch_bounds__(256) void adj_transpose_kernel(
    const float* __restrict__ adj, float* __restrict__ adjT) {
  __shared__ float tile[32][33];  // +1 pad: no LDS bank conflicts
  const int bx = blockIdx.x * 32, by = blockIdx.y * 32;
  const int tx = threadIdx.x, ty = threadIdx.y;  // 32 x 8
#pragma unroll
  for (int r = 0; r < 32; r += 8)
    tile[ty + r][tx] = adj[(by + ty + r) * N_NODES + (bx + tx)];
  __syncthreads();
#pragma unroll
  for (int r = 0; r < 32; r += 8)
    adjT[(bx + ty + r) * N_NODES + (by + tx)] = tile[tx][ty + r];
}

// ---------------------------------------------------------------------------
// Kernel 1: one-time W prep: transpose (1152x256 -> 256x1152) and split each
// f32 into f16 hi/lo pair (w ~= wh + wl, |err| ~ 2^-22 relative).
// ---------------------------------------------------------------------------
__global__ __launch_bounds__(256) void wprep_kernel(
    const float* __restrict__ W, _Float16* __restrict__ WhT,
    _Float16* __restrict__ WlT) {
  __shared__ float tile[32][33];
  const int bc = blockIdx.x * 32;  // contraction index base (36 blocks)
  const int bo = blockIdx.y * 32;  // out-channel base     (8 blocks)
  const int tx = threadIdx.x, ty = threadIdx.y;  // 32 x 8
#pragma unroll
  for (int r = 0; r < 32; r += 8)
    tile[ty + r][tx] = W[(bc + ty + r) * OUT_CH + (bo + tx)];
  __syncthreads();
#pragma unroll
  for (int r = 0; r < 32; r += 8) {
    const int c = bc + tx, o = bo + ty + r;
    const float v = tile[tx][ty + r];
    const _Float16 h = (_Float16)v;
    WhT[o * KF + c] = h;
    WlT[o * KF + c] = (_Float16)(v - (float)h);
  }
}

// ---------------------------------------------------------------------------
// Kernel 2: fused broadcast-product + top-9 selection.
// Block = node j (1024 blocks), thread = feature f (128 threads).
// Emits the GEMM A matrix directly as f16 hi/lo: Ah/Al[j, k*F + f].
// ---------------------------------------------------------------------------
__global__ __launch_bounds__(F_IN) void topk_kernel(
    const float* __restrict__ adjT, const float* __restrict__ fea,
    _Float16* __restrict__ Ah, _Float16* __restrict__ Al) {
  __shared__ float col[N_NODES];  // 4 KB: column j of adj
  const int j = blockIdx.x;
  const int f = threadIdx.x;
#pragma unroll
  for (int i = f; i < N_NODES; i += F_IN)  // coalesced (adjT row-contiguous)
    col[i] = adjT[j * N_NODES + i];
  __syncthreads();

  float t[KTOP];
#pragma unroll
  for (int k = 0; k < KTOP; ++k) t[k] = -3.4e38f;

  for (int i = 0; i < N_NODES; ++i) {
    const float v = col[i] * fea[i * F_IN + f];  // fea read coalesced over f
    if (v > t[KTOP - 1]) {
      // insert at tail; one bubble pass restores descending order
      t[KTOP - 1] = v;
#pragma unroll
      for (int k = KTOP - 1; k > 0; --k) {
        const float hi = fmaxf(t[k - 1], t[k]);
        const float lo = fminf(t[k - 1], t[k]);
        t[k - 1] = hi;
        t[k] = lo;
      }
    }
  }
#pragma unroll
  for (int k = 0; k < KTOP; ++k) {
    const float v = t[k];
    const _Float16 h = (_Float16)v;
    Ah[j * KF + k * F_IN + f] = h;                      // coalesced
    Al[j * KF + k * F_IN + f] = (_Float16)(v - (float)h);
  }
}

// ---------------------------------------------------------------------------
// Kernel 3: out = relu(A (1024x1152) * W (1152x256) + b) via WMMA.
// One wave32 per 16-row M-strip x 64 columns (4 N-subtiles, 4 accumulators):
// A fragment loaded once per K-step, reused by 12 v_wmma_f32_16x16x32_f16.
// Loads use uniform SGPR base + 32-bit per-lane element offsets so the
// backend emits saddr-form global_load_b128 with the kb term folded into the
// 24-bit instruction offset (no 64-bit VGPR address arithmetic in the loop).
// Fragment layout per ISA 16-bit 16x32: lanes 0-15 hold K {0..7,16..23},
// lanes 16-31 hold K {8..15,24..31} -> two 8-half groups at +0 and +16.
// ---------------------------------------------------------------------------
__global__ __launch_bounds__(128) void wmma_gemm_bias_relu_kernel(
    const _Float16* __restrict__ Ah, const _Float16* __restrict__ Al,
    const _Float16* __restrict__ WhT, const _Float16* __restrict__ WlT,
    const float* __restrict__ bias, float* __restrict__ out) {
  const int lane  = threadIdx.x & 31;
  const int wave  = threadIdx.x >> 5;          // n-group 0..3 (64 cols each)
  const int tm    = blockIdx.x * 16;           // 64 blocks: M-strip base
  const int l15   = lane & 15;
  const int khalf = (lane < 16) ? 0 : 8;

  // per-lane 32-bit element offsets (uniform base stays in SGPRs)
  const int aoff = (tm + l15) * KF + khalf;
  int boff[4];
#pragma unroll
  for (int s = 0; s < 4; ++s)
    boff[s] = (wave * 64 + s * 16 + l15) * KF + khalf;

  v8f acc[4] = {};
  for (int kb = 0; kb < KF; kb += 32) {
    const v8h a0h = *(const v8h*)&Ah[aoff + kb];
    const v8h a1h = *(const v8h*)&Ah[aoff + kb + 16];
    const v8h a0l = *(const v8h*)&Al[aoff + kb];
    const v8h a1l = *(const v8h*)&Al[aoff + kb + 16];
    const v16h ah = __builtin_shufflevector(a0h, a1h, 0, 1, 2, 3, 4, 5, 6, 7,
                                            8, 9, 10, 11, 12, 13, 14, 15);
    const v16h al = __builtin_shufflevector(a0l, a1l, 0, 1, 2, 3, 4, 5, 6, 7,
                                            8, 9, 10, 11, 12, 13, 14, 15);
#pragma unroll
    for (int s = 0; s < 4; ++s) {
      const v8h b0h = *(const v8h*)&WhT[boff[s] + kb];
      const v8h b1h = *(const v8h*)&WhT[boff[s] + kb + 16];
      const v8h b0l = *(const v8h*)&WlT[boff[s] + kb];
      const v8h b1l = *(const v8h*)&WlT[boff[s] + kb + 16];
      const v16h bh = __builtin_shufflevector(b0h, b1h, 0, 1, 2, 3, 4, 5, 6, 7,
                                              8, 9, 10, 11, 12, 13, 14, 15);
      const v16h bl = __builtin_shufflevector(b0l, b1l, 0, 1, 2, 3, 4, 5, 6, 7,
                                              8, 9, 10, 11, 12, 13, 14, 15);
      acc[s] = __builtin_amdgcn_wmma_f32_16x16x32_f16(false, al, false, bh,
                                                      (short)0, acc[s], false,
                                                      false);
      acc[s] = __builtin_amdgcn_wmma_f32_16x16x32_f16(false, ah, false, bl,
                                                      (short)0, acc[s], false,
                                                      false);
      acc[s] = __builtin_amdgcn_wmma_f32_16x16x32_f16(false, ah, false, bh,
                                                      (short)0, acc[s], false,
                                                      false);
    }
  }

  // C/D layout: VGPR r holds (M = r + (lane<16?0:8), N = lane&15)
#pragma unroll
  for (int s = 0; s < 4; ++s) {
    const int ncol = wave * 64 + s * 16 + l15;
    const float bn = bias[ncol];
#pragma unroll
    for (int r = 0; r < 8; ++r) {
      const int m = tm + r + ((lane < 16) ? 0 : 8);
      out[m * OUT_CH + ncol] = fmaxf(acc[s][r] + bn, 0.0f);
    }
  }
}

// ---------------------------------------------------------------------------
extern "C" void kernel_launch(void* const* d_in, const int* in_sizes, int n_in,
                              void* d_out, int out_size, void* d_ws,
                              size_t ws_size, hipStream_t stream) {
  const float* adj = (const float*)d_in[0];  // (1024,1024)
  const float* fea = (const float*)d_in[1];  // (1024,128)
  const float* W   = (const float*)d_in[2];  // (9,128,256) == (1152,256)
  const float* b   = (const float*)d_in[3];  // (256,)
  float* out = (float*)d_out;                // (1024,256)

  // workspace partition (all chunks well-aligned)
  float*    adjT = (float*)d_ws;                         // 4.00 MB
  _Float16* Ah   = (_Float16*)(adjT + (size_t)N_NODES * N_NODES);
  _Float16* Al   = Ah + (size_t)N_NODES * KF;            // 2.30 MB each
  _Float16* WhT  = Al + (size_t)N_NODES * KF;
  _Float16* WlT  = WhT + (size_t)OUT_CH * KF;            // 0.59 MB each

  dim3 tb(32, 8);
  adj_transpose_kernel<<<dim3(N_NODES / 32, N_NODES / 32), tb, 0, stream>>>(
      adj, adjT);
  wprep_kernel<<<dim3(KF / 32, OUT_CH / 32), tb, 0, stream>>>(W, WhT, WlT);
  topk_kernel<<<N_NODES, F_IN, 0, stream>>>(adjT, fea, Ah, Al);
  // 64 M-strips, 4 waves/block each covering 64 columns
  wmma_gemm_bias_relu_kernel<<<N_NODES / 16, 128, 0, stream>>>(Ah, Al, WhT,
                                                               WlT, b, out);
}
